// MultiHeadAttention_17025250361616
// MI455X (gfx1250) — matrix operand order repaired and checked by reading, not compiled, not run on hardware
//
#include <hip/hip_runtime.h>
#include <hip/hip_bf16.h>

#define NUM_HEADS 16
#define D_MODEL 2048
#define HEAD_DIM 128
#define SEQ 2048
#define BATCH 2

typedef _Float16 h8   __attribute__((ext_vector_type(8)));
typedef _Float16 v16h __attribute__((ext_vector_type(16)));
typedef float    v8f  __attribute__((ext_vector_type(8)));

union FragU { v16h v; h8 h[2]; };

// A operand (16x32, M x K), from row-major f16 [rows][ld], K contiguous.
// Lane: m = lane%16, half = lane/16. elems 0..7 -> K = k0+8*half+e,
// elems 8..15 -> K = k0+16+8*half+(e-8).  (ISA 7.12.2, 16-bit A 16x32)
__device__ __forceinline__ v16h load_frag_a(const _Float16* base, int ld, int row0, int k0) {
    const int lane = threadIdx.x & 31;
    const int m = lane & 15, hf = lane >> 4;
    const _Float16* p = base + (size_t)(row0 + m) * ld + k0 + hf * 8;
    FragU u;
    u.h[0] = *(const h8*)(p);
    u.h[1] = *(const h8*)(p + 16);
    return u.v;
}

// B operand (32x16, K x N) loaded from TRANSPOSED storage Bt[n][k] (K contiguous).
// Lane: n = lane%16, half = lane/16. elems 0..7 -> K = k0+8*half+e,
// elems 8..15 -> K = k0+16+8*half+(e-8).
__device__ __forceinline__ v16h load_frag_bt(const _Float16* bt, int ld, int n0, int k0) {
    const int lane = threadIdx.x & 31;
    const int n = lane & 15, hf = lane >> 4;
    const _Float16* p = bt + (size_t)(n0 + n) * ld + k0 + hf * 8;
    FragU u;
    u.h[0] = *(const h8*)(p);
    u.h[1] = *(const h8*)(p + 16);
    return u.v;
}

__device__ __forceinline__ v8f wmma16(v16h a, v16h b, v8f c) {
    return __builtin_amdgcn_wmma_f32_16x16x32_f16(false, a, false, b, (short)0, c,
                                                  false, false);
}

// ---------------- elementwise conversions ----------------
__global__ void k_cvt(const float* __restrict__ in, _Float16* __restrict__ out, long n) {
    long i = (long)blockIdx.x * blockDim.x + threadIdx.x;
    long st = (long)gridDim.x * blockDim.x;
    for (; i < n; i += st) out[i] = (_Float16)in[i];
}

// out[z][c][r] = in[z][r][c]  (f32 -> f16), square transpose+convert
__global__ void k_tcvt(const float* __restrict__ in, _Float16* __restrict__ out,
                       int rows, int cols) {
    const long nb = (long)rows * cols;
    const float* ib = in + (size_t)blockIdx.z * nb;
    _Float16* ob = out + (size_t)blockIdx.z * nb;
    long i = (long)blockIdx.x * blockDim.x + threadIdx.x;
    long st = (long)gridDim.x * blockDim.x;
    for (; i < nb; i += st) {
        int r = (int)(i % rows);
        int c = (int)(i / rows);
        ob[i] = (_Float16)ib[(long)r * cols + c];
    }
}

// ---------------- QKV GEMM + RoPE ----------------
// grid: (BS/128, NUM_HEADS, 3), block 128 (4 waves).
// Block tile: 128 rows x 128 cols (one head); wave: 32 rows x 128 cols.
__global__ __launch_bounds__(128) void k_qkv_rope(
        const _Float16* __restrict__ x16, const _Float16* __restrict__ wqkv_t,
        _Float16* __restrict__ q16, _Float16* __restrict__ k16,
        _Float16* __restrict__ v16t) {
    const int tile = blockIdx.x;
    const int h = blockIdx.y;
    const int t = blockIdx.z;
    const int w = threadIdx.x >> 5;
    const int lane = threadIdx.x & 31;
    const int n = lane & 15, hf = lane >> 4;

    __shared__ float tileC[128][HEAD_DIM + 8];

    const _Float16* wt = wqkv_t + (size_t)t * D_MODEL * D_MODEL;
    const int row0 = tile * 128 + w * 32;
    const int colbase = h * HEAD_DIM;

    const v8f vzero = {0.f,0.f,0.f,0.f,0.f,0.f,0.f,0.f};
    v8f acc[2][8];
#pragma unroll
    for (int s2 = 0; s2 < 2; ++s2)
#pragma unroll
        for (int j = 0; j < 8; ++j) acc[s2][j] = vzero;

    for (int kk = 0; kk < D_MODEL; kk += 32) {
        v16h a0 = load_frag_a(x16, D_MODEL, row0, kk);
        v16h a1 = load_frag_a(x16, D_MODEL, row0 + 16, kk);
#pragma unroll
        for (int j = 0; j < 8; ++j) {
            v16h bf = load_frag_bt(wt, D_MODEL, colbase + j * 16, kk);
            acc[0][j] = wmma16(a0, bf, acc[0][j]);
            acc[1][j] = wmma16(a1, bf, acc[1][j]);
        }
    }
#pragma unroll
    for (int s2 = 0; s2 < 2; ++s2)
#pragma unroll
        for (int j = 0; j < 8; ++j)
#pragma unroll
            for (int r = 0; r < 8; ++r)
                tileC[w * 32 + s2 * 16 + r + 8 * hf][j * 16 + n] = acc[s2][j][r];
    __syncthreads();

    for (int idx = threadIdx.x; idx < 128 * HEAD_DIM; idx += blockDim.x) {
        const int lr = idx >> 7;
        const int col = idx & 127;
        const int bs = tile * 128 + lr;
        const int bb = bs / SEQ, s = bs % SEQ;
        const size_t bh = (size_t)bb * NUM_HEADS + h;
        if (t == 2) {
            // store V transposed: [b,h,d,s]
            v16t[(bh * HEAD_DIM + col) * SEQ + s] = (_Float16)tileC[lr][col];
        } else {
            const int j = col & 63;
            const float inv_freq = __powf(10000.0f, -(float)j * (1.0f / 64.0f));
            float sn, cs;
            __sincosf((float)s * inv_freq, &sn, &cs);
            const float t0 = tileC[lr][2 * j];
            const float t1 = tileC[lr][2 * j + 1];
            const float v = (col < 64) ? (t0 * cs - t1 * sn) : (t0 * sn + t1 * cs);
            _Float16* dst = (t == 0) ? q16 : k16;
            dst[(bh * SEQ + s) * HEAD_DIM + col] = (_Float16)v;
        }
    }
}

// ---------------- flash attention ----------------
// grid: (SEQ/64, NUM_HEADS, BATCH), block 128 (4 waves); wave handles 16 queries.
__global__ __launch_bounds__(128) void k_flash(
        const _Float16* __restrict__ q16, const _Float16* __restrict__ k16,
        const _Float16* __restrict__ v16t, const int* __restrict__ attn_mask,
        _Float16* __restrict__ attn16) {
    const int qt = blockIdx.x;
    const int h = blockIdx.y;
    const int bb = blockIdx.z;
    const int w = threadIdx.x >> 5;
    const int lane = threadIdx.x & 31;
    const int n = lane & 15, hf = lane >> 4;

    __shared__ __align__(16) _Float16 pS[4][16][32];

    const size_t bh = (size_t)bb * NUM_HEADS + h;
    const _Float16* qb   = q16  + bh * SEQ * HEAD_DIM;
    const _Float16* kmat = k16  + bh * SEQ * HEAD_DIM;
    const _Float16* vb   = v16t + bh * HEAD_DIM * SEQ;
    const int* mk = attn_mask + (size_t)bb * SEQ;

    const int q0 = qt * 64 + w * 16;

    v16h aq[4];
#pragma unroll
    for (int kk = 0; kk < 4; ++kk) aq[kk] = load_frag_a(qb, HEAD_DIM, q0, kk * 32);

    const v8f vzero = {0.f,0.f,0.f,0.f,0.f,0.f,0.f,0.f};
    v8f o[8];
#pragma unroll
    for (int j = 0; j < 8; ++j) o[j] = vzero;
    float mrow[8], lrow[8];
#pragma unroll
    for (int r = 0; r < 8; ++r) { mrow[r] = -1e30f; lrow[r] = 0.0f; }

    const float scale = 0.08838834764831845f;  // 1/sqrt(128)
    const int kmax = qt * 64 + 64;             // block-uniform causal bound

    for (int kb = 0; kb < kmax; kb += 32) {
        v8f s0 = vzero, s1 = vzero;
#pragma unroll
        for (int kk = 0; kk < 4; ++kk) {
            v16h b0 = load_frag_bt(kmat, HEAD_DIM, kb, kk * 32);
            v16h b1 = load_frag_bt(kmat, HEAD_DIM, kb + 16, kk * 32);
            s0 = wmma16(aq[kk], b0, s0);
            s1 = wmma16(aq[kk], b1, s1);
        }
        const int key0 = kb + n, key1 = kb + 16 + n;
        const int km0 = mk[key0], km1 = mk[key1];
#pragma unroll
        for (int r = 0; r < 8; ++r) {
            const int q = q0 + r + 8 * hf;
            const bool a0 = (key0 == q) || (key0 < q && km0 != 0);
            const bool a1 = (key1 == q) || (key1 < q && km1 != 0);
            const float v0 = s0[r] * scale + (a0 ? 0.0f : -1e9f);
            const float v1 = s1[r] * scale + (a1 ? 0.0f : -1e9f);
            float mx = fmaxf(v0, v1);
            mx = fmaxf(mx, __shfl_xor(mx, 1, 32));
            mx = fmaxf(mx, __shfl_xor(mx, 2, 32));
            mx = fmaxf(mx, __shfl_xor(mx, 4, 32));
            mx = fmaxf(mx, __shfl_xor(mx, 8, 32));
            const float mnew = fmaxf(mrow[r], mx);
            const float sc = __expf(mrow[r] - mnew);
            const float p0 = __expf(v0 - mnew);
            const float p1 = __expf(v1 - mnew);
            float rs = p0 + p1;
            rs += __shfl_xor(rs, 1, 32);
            rs += __shfl_xor(rs, 2, 32);
            rs += __shfl_xor(rs, 4, 32);
            rs += __shfl_xor(rs, 8, 32);
            lrow[r] = lrow[r] * sc + rs;
            mrow[r] = mnew;
#pragma unroll
            for (int j = 0; j < 8; ++j) o[j][r] *= sc;
            pS[w][r + 8 * hf][n]      = (_Float16)p0;
            pS[w][r + 8 * hf][16 + n] = (_Float16)p1;
        }
        __syncthreads();  // order LDS P writes before cross-lane reads (uniform trip count)
        v16h ap = load_frag_a(&pS[w][0][0], 32, 0, 0);
#pragma unroll
        for (int j = 0; j < 8; ++j) {
            v16h bv = load_frag_bt(vb, SEQ, j * 16, kb);  // V^T: dims rows, keys contiguous
            o[j] = wmma16(ap, bv, o[j]);
        }
        __syncthreads();
    }

#pragma unroll
    for (int j = 0; j < 8; ++j)
#pragma unroll
        for (int r = 0; r < 8; ++r) {
            const int q = q0 + r + 8 * hf;
            const float val = o[j][r] * (1.0f / lrow[r]);
            attn16[((size_t)bb * SEQ + q) * D_MODEL + h * HEAD_DIM + j * 16 + n] =
                (_Float16)val;
        }
}

// ---------------- out projection ----------------
// grid: (BS/128, D_MODEL/128), block 128 (4 waves).
// Block tile: 128 rows x 128 cols; wave: 32 rows x 128 cols.
__global__ __launch_bounds__(128) void k_outproj(
        const _Float16* __restrict__ attn16, const _Float16* __restrict__ wo_t,
        float* __restrict__ out) {
    const int tr = blockIdx.x;
    const int tc = blockIdx.y;
    const int w = threadIdx.x >> 5;
    const int lane = threadIdx.x & 31;
    const int n = lane & 15, hf = lane >> 4;
    const int row0 = tr * 128 + w * 32;
    const int col0 = tc * 128;

    const v8f vzero = {0.f,0.f,0.f,0.f,0.f,0.f,0.f,0.f};
    v8f acc[2][8];
#pragma unroll
    for (int s2 = 0; s2 < 2; ++s2)
#pragma unroll
        for (int j = 0; j < 8; ++j) acc[s2][j] = vzero;

    for (int kk = 0; kk < D_MODEL; kk += 32) {
        v16h a0 = load_frag_a(attn16, D_MODEL, row0, kk);
        v16h a1 = load_frag_a(attn16, D_MODEL, row0 + 16, kk);
#pragma unroll
        for (int j = 0; j < 8; ++j) {
            v16h bf = load_frag_bt(wo_t, D_MODEL, col0 + j * 16, kk);
            acc[0][j] = wmma16(a0, bf, acc[0][j]);
            acc[1][j] = wmma16(a1, bf, acc[1][j]);
        }
    }
#pragma unroll
    for (int s2 = 0; s2 < 2; ++s2)
#pragma unroll
        for (int j = 0; j < 8; ++j)
#pragma unroll
            for (int r = 0; r < 8; ++r)
                out[(size_t)(row0 + s2 * 16 + r + 8 * hf) * D_MODEL + col0 + j * 16 + n] =
                    acc[s2][j][r];
}

extern "C" void kernel_launch(void* const* d_in, const int* in_sizes, int n_in,
                              void* d_out, int out_size, void* d_ws, size_t ws_size,
                              hipStream_t stream) {
    (void)in_sizes; (void)n_in; (void)out_size; (void)ws_size;
    const float* x        = (const float*)d_in[0];
    const int*   attn_msk = (const int*)d_in[1];
    const float* in_proj  = (const float*)d_in[2];
    const float* out_proj = (const float*)d_in[3];
    float* out = (float*)d_out;

    const size_t BS = (size_t)BATCH * SEQ;          // 4096
    const size_t MAT = BS * D_MODEL;                // 8,388,608 elements

    _Float16* ws     = (_Float16*)d_ws;
    _Float16* x16    = ws;                                     // [BS][D]
    _Float16* wqkv_t = x16 + MAT;                              // [3][col][d]
    _Float16* wo_t   = wqkv_t + (size_t)3 * D_MODEL * D_MODEL; // [o][hd]
    _Float16* q16    = wo_t + (size_t)D_MODEL * D_MODEL;       // [b,h,s,d]
    _Float16* k16    = q16 + MAT;                              // [b,h,s,d]
    _Float16* v16t   = k16 + MAT;                              // [b,h,d,s]
    _Float16* attn16 = v16t + MAT;                             // [bs][h*d]

    k_cvt<<<1024, 256, 0, stream>>>(x, x16, (long)MAT);
    k_tcvt<<<dim3(512, 1, 3), 256, 0, stream>>>(in_proj, wqkv_t, D_MODEL, D_MODEL);
    k_tcvt<<<dim3(512, 1, 1), 256, 0, stream>>>(out_proj, wo_t, D_MODEL, D_MODEL);
    k_qkv_rope<<<dim3(BS / 128, NUM_HEADS, 3), 128, 0, stream>>>(x16, wqkv_t, q16, k16, v16t);
    k_flash<<<dim3(SEQ / 64, NUM_HEADS, BATCH), 128, 0, stream>>>(q16, k16, v16t, attn_msk, attn16);
    k_outproj<<<dim3(BS / 128, D_MODEL / 128), 128, 0, stream>>>(attn16, wo_t, out);
}
